// APPNP_65919158059080
// MI455X (gfx1250) — matrix operand order repaired and checked by reading, not compiled, not run on hardware
//
#include <hip/hip_runtime.h>
#include <hip/hip_bf16.h>
#include <stdint.h>
#include <stddef.h>

// ---------------------------------------------------------------------------
// APPNP on MI455X (gfx1250):
//   h1 = relu(x@W1+b1)   : bf16 WMMA (v_wmma_f32_16x16x32_bf16), f32 accum
//   h  = h1@W2+b2        : bf16 WMMA, f32 accum
//   10x: h = (1-a)*spmm + a*h : L2-resident gather + fp32 global atomics
//
// GEMM: double-buffered LDS staging, K-step 64 (2 chained WMMAs per barrier
// period), global_prefetch for the K+128 tile of A, native packed f32->bf16
// conversion. N and K are template constants so all strided/unrolled accesses
// lower to immediate-offset loads/stores (no per-element 64-bit muls).
// ---------------------------------------------------------------------------

#define APPNP_ALPHA 0.01f

typedef __attribute__((ext_vector_type(16))) __bf16 v16bf;
typedef __attribute__((ext_vector_type(8)))  float  v8f;

static __device__ __forceinline__ uint32_t pack_bf16x2(float lo, float hi) {
  union { __hip_bfloat162 h; uint32_t u; } c;
  c.h = __float22bfloat162_rn(make_float2(lo, hi));   // native packed cvt
  return c.u;
}
static __device__ __forceinline__ unsigned short f32_to_bf16_bits(float f) {
  union { __hip_bfloat16 h; unsigned short u; } c;
  c.h = __float2bfloat16(f);                          // native cvt (RNE)
  return c.u;
}

// LDS row stride in halves: 64 K-elements + 16 pad (160B rows, 16B aligned,
// odd multiple of 32B so b128 fragment loads spread across banks).
#define LDS_STRIDE 80

// ---------------------------------------------------------------------------
// Tiled GEMM: Out[M,N] = act(A[M,K] @ W[K,N] + bias[N])
// Block tile 32(M) x 64(N), 8 waves (2x4), one 16x16 WMMA tile per wave.
// K-step 64: A staged row-major bf16, W staged column-major bf16; fragments
// are two ds_load_b128 each (per the 16-bit A 16x32 / B 32x16 VGPR layouts).
// ---------------------------------------------------------------------------
template <int N, int K, bool A_IS_BF16, bool RELU, bool OUT_IS_BF16>
__global__ __launch_bounds__(256) void wmma_gemm_bias_act(
    const void* __restrict__ Av, const float* __restrict__ W,
    const float* __restrict__ bias, void* __restrict__ Outv)
{
  __shared__ __align__(16) unsigned short lA[2][32 * LDS_STRIDE];  // 32 rows x 64 K
  __shared__ __align__(16) unsigned short lB[2][64 * LDS_STRIDE];  // 64 cols x 64 K (col-major)

  const int t    = threadIdx.x;
  const int lane = t & 31;
  const int wave = t >> 5;
  const int wm   = (wave & 1) * 16;   // wave M offset in block tile
  const int wn   = (wave >> 1) * 16;  // wave N offset in block tile
  const int rowBase = blockIdx.x * 32;
  const int colBase = blockIdx.y * 64;

  const int half16 = lane >> 4;       // which half-wave (K striping select)

  // Staging work split.
  const int am  = t >> 3;             // 0..31 (A row), 8 k-elems each
  const int ak  = (t & 7) * 8;        // 0..56
  const int bn  = t & 63;             // 0..63 (B col), 16 k-elems each
  const int bkb = (t >> 6) * 16;      // 0,16,32,48

  // Hoisted base pointers (one 64-bit address computation each).
  constexpr int ESZ = A_IS_BF16 ? 2 : 4;
  const char*  aBase  = (const char*)Av + ((size_t)(rowBase + am) * K + ak) * ESZ;
  const float* wBase  = W + (size_t)bkb * N + (colBase + bn);
  unsigned short* laRow = &lA[0][am * LDS_STRIDE + ak];
  unsigned short* lbCol = &lB[0][bn * LDS_STRIDE + bkb];

  v8f acc = {};

  // ---- staging helpers: constant strides -> immediate-offset accesses ----
  auto stageA = [&](int buf, int kk) {
    unsigned short* dst = laRow + buf * (32 * LDS_STRIDE);
    if constexpr (A_IS_BF16) {
      const uint4 v = *(const uint4*)(aBase + (size_t)kk * 2);
      *(uint4*)dst = v;
    } else {
      const float* p = (const float*)(aBase + (size_t)kk * 4);
      const float4 f0 = *(const float4*)(p);
      const float4 f1 = *(const float4*)(p + 4);
      uint4 v;
      v.x = pack_bf16x2(f0.x, f0.y);
      v.y = pack_bf16x2(f0.z, f0.w);
      v.z = pack_bf16x2(f1.x, f1.y);
      v.w = pack_bf16x2(f1.z, f1.w);
      *(uint4*)dst = v;
    }
  };
  auto stageB = [&](int buf, int kk) {
    const float* wp = wBase + (size_t)kk * N;   // 16 loads at constant i*N offsets
    float f[16];
#pragma unroll
    for (int i = 0; i < 16; ++i)
      f[i] = wp[i * N];
    uint4 p0, p1;
    p0.x = pack_bf16x2(f[0],  f[1]);  p0.y = pack_bf16x2(f[2],  f[3]);
    p0.z = pack_bf16x2(f[4],  f[5]);  p0.w = pack_bf16x2(f[6],  f[7]);
    p1.x = pack_bf16x2(f[8],  f[9]);  p1.y = pack_bf16x2(f[10], f[11]);
    p1.z = pack_bf16x2(f[12], f[13]); p1.w = pack_bf16x2(f[14], f[15]);
    unsigned short* dst = lbCol + buf * (64 * LDS_STRIDE);
    *(uint4*)(dst)     = p0;
    *(uint4*)(dst + 8) = p1;
  };

  // Prologue: stage first K-tile.
  stageA(0, 0);
  stageB(0, 0);

  int buf = 0;
  for (int kk = 0; kk < K; kk += 64) {
    __syncthreads();

    // Prefetch A two tiles ahead (global_prefetch_b8).
    if (kk + 128 < K)
      __builtin_prefetch(aBase + (size_t)(kk + 128) * ESZ, 0, 1);

    // Stage next tile into the other buffer while computing this one.
    if (kk + 64 < K) {
      stageA(buf ^ 1, kk + 64);
      stageB(buf ^ 1, kk + 64);
    }

    // Two chained WMMAs over the 64-wide K tile.
    const unsigned short* aRowPtr = &lA[buf][(wm + (lane & 15)) * LDS_STRIDE];
    const unsigned short* bColPtr = &lB[buf][(wn + (lane & 15)) * LDS_STRIDE];
#pragma unroll
    for (int ks = 0; ks < 64; ks += 32) {
      union { uint4 q[2]; v16bf v; } ua, ub;
      // A 16x32 bf16 layout: half-wave k-chunks at +0/+16 halves, +16B select.
      ua.q[0] = *(const uint4*)(aRowPtr + ks + half16 * 8);
      ua.q[1] = *(const uint4*)(aRowPtr + ks + half16 * 8 + 16);
      // B 32x16 bf16 layout: half-wave k-chunks at +0/+16 halves, +16B select.
      ub.q[0] = *(const uint4*)(bColPtr + ks + half16 * 8);
      ub.q[1] = *(const uint4*)(bColPtr + ks + half16 * 8 + 16);
      acc = __builtin_amdgcn_wmma_f32_16x16x32_bf16(
          /*neg_a=*/false, ua.v, /*neg_b=*/false, ub.v,
          /*c_mod=*/(short)0, acc, /*reuse_a=*/false, /*reuse_b=*/false);
    }
    buf ^= 1;
  }

  // Epilogue: bias + optional relu; C/D layout: VGPR r -> row r + 8*(lane>>4),
  // col = lane&15.  Base pointer hoisted; r*N are immediate store offsets.
  const int outCol = colBase + wn + (lane & 15);
  const int row0   = rowBase + wm + half16 * 8;
  const float bv = bias[outCol];
  if constexpr (OUT_IS_BF16) {
    unsigned short* op = (unsigned short*)Outv + (size_t)row0 * N + outCol;
#pragma unroll
    for (int r = 0; r < 8; ++r) {
      float v = acc[r] + bv;
      if (RELU) v = fmaxf(v, 0.0f);
      op[r * N] = f32_to_bf16_bits(v);
    }
  } else {
    float* op = (float*)Outv + (size_t)row0 * N + outCol;
#pragma unroll
    for (int r = 0; r < 8; ++r) {
      float v = acc[r] + bv;
      if (RELU) v = fmaxf(v, 0.0f);
      op[r * N] = v;
    }
  }
}

// ---------------------------------------------------------------------------
// Propagation: dst = ALPHA * cur  (init), then edges atomically add
// (1-ALPHA)*val[e]*cur[col[e]] into dst[row[e]].  h (25.6MB) is L2-resident.
// ---------------------------------------------------------------------------
__global__ __launch_bounds__(256) void axpy_init_kernel(
    const float* __restrict__ H, float* __restrict__ D, int n4)
{
  int i = blockIdx.x * blockDim.x + threadIdx.x;
  if (i < n4) {
    const float4 v = ((const float4*)H)[i];
    float4 o;
    o.x = APPNP_ALPHA * v.x; o.y = APPNP_ALPHA * v.y;
    o.z = APPNP_ALPHA * v.z; o.w = APPNP_ALPHA * v.w;
    ((float4*)D)[i] = o;
  }
}

__global__ __launch_bounds__(256) void edge_spmm_kernel(
    const int* __restrict__ row, const int* __restrict__ col,
    const float* __restrict__ val,
    const float* __restrict__ Hsrc, float* __restrict__ Hdst, int E)
{
  int tid = blockIdx.x * blockDim.x + threadIdx.x;
  int e = tid >> 2;              // 4 threads per edge
  if (e >= E) return;
  const int fc = (tid & 3) * 16; // each thread: 16 features
  const int r = row[e];
  const int c = col[e];
  const float w = val[e] * (1.0f - APPNP_ALPHA);
  const float4* src = (const float4*)(Hsrc + (size_t)c * 64 + fc);
  float* dst = Hdst + (size_t)r * 64 + fc;
#pragma unroll
  for (int j = 0; j < 4; ++j) {
    const float4 v = src[j];
    atomicAdd(dst + j * 4 + 0, w * v.x);
    atomicAdd(dst + j * 4 + 1, w * v.y);
    atomicAdd(dst + j * 4 + 2, w * v.z);
    atomicAdd(dst + j * 4 + 3, w * v.w);
  }
}

// ---------------------------------------------------------------------------
extern "C" void kernel_launch(void* const* d_in, const int* in_sizes, int n_in,
                              void* d_out, int out_size, void* d_ws, size_t ws_size,
                              hipStream_t stream)
{
  const float* x   = (const float*)d_in[0];
  const int*   row = (const int*)  d_in[1];
  const int*   col = (const int*)  d_in[2];
  const float* val = (const float*)d_in[3];
  const float* W1  = (const float*)d_in[4];
  const float* b1  = (const float*)d_in[5];
  const float* W2  = (const float*)d_in[6];
  const float* b2  = (const float*)d_in[7];

  const int IN_F = 512, HID_F = 256, OUT_F = 64, KITERS = 10;
  const int M = in_sizes[0] / IN_F;   // 100000 (multiple of 32)
  const int E = in_sizes[1];          // 3200000

  // Workspace layout: h1 (bf16, M*HID) then H0 (f32, M*OUT).
  // H1 aliases h1's storage (h1 is dead once the second GEMM finishes).
  unsigned short* h1 = (unsigned short*)d_ws;
  float* H0 = (float*)((char*)d_ws + (size_t)M * HID_F * sizeof(unsigned short));
  float* H1 = (float*)d_ws;

  dim3 blk(256);

  // h1 = relu(x @ W1 + b1), bf16 output
  wmma_gemm_bias_act<256, 512, false, true, true>
      <<<dim3(M / 32, HID_F / 64), blk, 0, stream>>>(x, W1, b1, h1);

  // H0 = h1 @ W2 + b2, f32 output
  wmma_gemm_bias_act<64, 256, true, false, false>
      <<<dim3(M / 32, OUT_F / 64), blk, 0, stream>>>(h1, W2, b2, H0);

  // K rounds of APPNP propagation; final round writes directly to d_out.
  const int n4 = M * OUT_F / 4;
  const int initBlocks = (n4 + 255) / 256;
  const int edgeBlocks = (int)(((size_t)E * 4 + 255) / 256);

  float* cur = H0;
  for (int k = 0; k < KITERS; ++k) {
    float* dst = (k == KITERS - 1) ? (float*)d_out : ((k & 1) ? H0 : H1);
    axpy_init_kernel<<<initBlocks, blk, 0, stream>>>(cur, dst, n4);
    edge_spmm_kernel<<<edgeBlocks, blk, 0, stream>>>(row, col, val, cur, dst, E);
    cur = dst;
  }
}
